// MultiScaleMultiHeadedAttention_38955353375581
// MI455X (gfx1250) — matrix operand order repaired and checked by reading, not compiled, not run on hardware
//
#include <hip/hip_runtime.h>

// ---------------- CDNA5 (gfx1250) types & helpers -------------------------

typedef __attribute__((ext_vector_type(16))) __bf16 v16bf;
typedef __attribute__((ext_vector_type(8)))  float  v8f;
typedef __attribute__((ext_vector_type(4)))  int    v4i;

struct __attribute__((aligned(16))) U128 { unsigned int a, b, c, d; };
struct __attribute__((aligned(16))) U256 { U128 lo, hi; };
struct __attribute__((aligned(16))) F128 { float x, y, z, w; };

static __device__ __forceinline__ unsigned short f2bf(float f) {
  unsigned u = __float_as_uint(f);
  u += 0x7FFFu + ((u >> 16) & 1u);          // round-to-nearest-even
  return (unsigned short)(u >> 16);
}

// Build a 16-element bf16 WMMA fragment from two 16B chunks (per ISA VGPR layout).
static __device__ __forceinline__ v16bf ld_frag(const unsigned short* p0,
                                                const unsigned short* p1) {
  U256 t;
  t.lo = *(const U128*)p0;
  t.hi = *(const U128*)p1;
  return __builtin_bit_cast(v16bf, t);
}

static __device__ __forceinline__ v8f wmma_bf16(v16bf a, v16bf b, v8f c) {
  return __builtin_amdgcn_wmma_f32_16x16x32_bf16(false, a, false, b,
                                                 (short)0, c, false, false);
}

// ---- async global->LDS staging (CDNA5 GLOBAL_LOAD_ASYNC_TO_LDS_B128) -----
#if __has_builtin(__builtin_amdgcn_global_load_async_to_lds_b128) && \
    __has_builtin(__builtin_amdgcn_s_wait_asynccnt)
#define USE_ASYNC_LDS 1
#else
#define USE_ASYNC_LDS 0
#endif

static __device__ __forceinline__ void copy16_g2s(unsigned short* lds,
                                                  const unsigned short* g) {
#if USE_ASYNC_LDS
  // Prototype (from hipcc diagnostic): (v4i AS1* src, v4i AS3* dst, imm, imm)
  __builtin_amdgcn_global_load_async_to_lds_b128(
      (v4i __attribute__((address_space(1))) *)(v4i*)g,
      (v4i __attribute__((address_space(3))) *)(v4i*)lds, 0, 0);
#else
  *(U128*)lds = *(const U128*)g;
#endif
}

static __device__ __forceinline__ void wait_g2s() {
#if USE_ASYNC_LDS
  __builtin_amdgcn_s_wait_asynccnt(0);
#endif
}

// ---------------- fp32 -> bf16 conversion ---------------------------------

__global__ void cvt_f32_bf16(const float* __restrict__ in,
                             unsigned short* __restrict__ out, int n) {
  int i = (blockIdx.x * blockDim.x + threadIdx.x) * 4;
  if (i + 3 < n) {
    F128 f = *(const F128*)(in + i);
    unsigned short o[4] = {f2bf(f.x), f2bf(f.y), f2bf(f.z), f2bf(f.w)};
    unsigned long long pack;
    __builtin_memcpy(&pack, o, 8);
    *(unsigned long long*)(out + i) = pack;
  }
}

// ---------------- generic WMMA GEMM ---------------------------------------
// C[M,N] = A[M,K](bf16) x B[K,N](bf16) + bias
// MODE: 0 = bf16 out, 1 = relu + bf16 out, 2 = f32 out, 3 = head-scatter bf16
// Block tile 128x128, BK=32, 256 threads (8 wave32s), wave tile 64x32.

#define GBM 128
#define GBN 128
#define GBK 32
#define LSTR 40   // LDS row stride in ushorts (padded, 16B aligned)

template <int MODE>
__global__ void __launch_bounds__(256)
gemm_wmma(const unsigned short* __restrict__ A, int lda,
          const unsigned short* __restrict__ Bw, int ldb,
          const float* __restrict__ bias,
          void* __restrict__ Cout, int ldc, int colOff,
          int M, int N, int K) {
  __shared__ __attribute__((aligned(16))) unsigned short aT[GBM * LSTR];
  __shared__ __attribute__((aligned(16))) unsigned short bT[GBN * LSTR];

  const int tid = threadIdx.x;
  const int lane = tid & 31;
  const int wave = tid >> 5;
  const int hi = lane >> 4;
  const int l16 = lane & 15;

  const int blockRow = blockIdx.y * GBM;
  const int blockCol = blockIdx.x * GBN;
  const int waveRow = (wave & 1) * 64;
  const int waveCol = (wave >> 1) * 32;

  v8f acc[4][2];
#pragma unroll
  for (int i = 0; i < 4; ++i)
#pragma unroll
    for (int g = 0; g < 2; ++g)
#pragma unroll
      for (int e = 0; e < 8; ++e) acc[i][g][e] = 0.0f;

  for (int k0 = 0; k0 < K; k0 += GBK) {
    __syncthreads();
    // Stage A tile 128x32 (row major), b128 per chunk.
#pragma unroll
    for (int it = 0; it < 2; ++it) {
      int c = tid * 2 + it;              // 0..511 chunks of 8 bf16
      int row = c >> 2;
      int kc = (c & 3) * 8;
      copy16_g2s(aT + row * LSTR + kc,
                 A + (size_t)(blockRow + row) * lda + k0 + kc);
    }
    // Stage B tile 32x128 transposed -> bT[n][k]
#pragma unroll
    for (int it = 0; it < 2; ++it) {
      int c = tid * 2 + it;              // 0..511
      int kr = c >> 4;                   // 0..31
      int nc = (c & 15) * 8;
      U128 d = *(const U128*)(Bw + (size_t)(k0 + kr) * ldb + blockCol + nc);
      const unsigned short* s16 = (const unsigned short*)&d;
#pragma unroll
      for (int j = 0; j < 8; ++j) bT[(nc + j) * LSTR + kr] = s16[j];
    }
    wait_g2s();
    __syncthreads();

    // A fragments: lane l16 = row M, hi half selects K 8..15 / 24..31.
    v16bf ar[4];
#pragma unroll
    for (int i = 0; i < 4; ++i) {
      const unsigned short* p = aT + (waveRow + i * 16 + l16) * LSTR;
      ar[i] = ld_frag(p + (hi ? 8 : 0), p + 16 + (hi ? 8 : 0));
    }
    // B fragments: lane l16 = col N, hi half selects K 8..15 / 24..31.
    v16bf br[2];
#pragma unroll
    for (int g = 0; g < 2; ++g) {
      const unsigned short* p = bT + (waveCol + g * 16 + l16) * LSTR + (hi ? 8 : 0);
      br[g] = ld_frag(p, p + 16);
    }
#pragma unroll
    for (int i = 0; i < 4; ++i)
#pragma unroll
      for (int g = 0; g < 2; ++g) acc[i][g] = wmma_bf16(ar[i], br[g], acc[i][g]);
  }

  // Epilogue: C layout = lane holds (M = r + 8*hi, N = l16) per 16x16 tile.
#pragma unroll
  for (int i = 0; i < 4; ++i) {
#pragma unroll
    for (int g = 0; g < 2; ++g) {
#pragma unroll
      for (int r = 0; r < 8; ++r) {
        int row = blockRow + waveRow + i * 16 + r + hi * 8;
        int col = blockCol + waveCol + g * 16 + l16;
        float v = acc[i][g][r] + bias[col];
        if (MODE == 1) v = fmaxf(v, 0.0f);
        if (MODE == 2) {
          ((float*)Cout)[(size_t)row * ldc + colOff + col] = v;
        } else if (MODE == 3) {
          // (row = b*512+n, col = h*64+d) -> [(b*16+h)*512+n]*64+d
          int b = row >> 9, n = row & 511;
          int hh = col >> 6, dd = col & 63;
          ((unsigned short*)Cout)[(((size_t)(b * 16 + hh) * 512 + n) * 64) + dd] =
              f2bf(v);
        } else {
          ((unsigned short*)Cout)[(size_t)row * ldc + colOff + col] = f2bf(v);
        }
      }
    }
  }
}

// ---------------- fused multi-threshold flash attention -------------------
// grid: (B*H = 256, N/64 = 8), block 128 (4 wave32s).
// Each wave owns a 16-row q tile; the 4 waves share K/V 32-key tiles in LDS.
// dist_conv computed inline; 3 online softmaxes (thr 0.3 / 0.6 / dense).

#define NEGV (-1e10f)

__global__ void __launch_bounds__(128)
attn_fused(const unsigned short* __restrict__ qB,
           const unsigned short* __restrict__ kB,
           const unsigned short* __restrict__ vB,
           const float* __restrict__ dist,
           const float* __restrict__ c1w, const float* __restrict__ c1b,
           const float* __restrict__ c2w, const float* __restrict__ c2b,
           unsigned short* __restrict__ out, size_t outStrideT) {
  __shared__ __attribute__((aligned(16))) unsigned short kT[32 * 64];      // [key][feat]
  __shared__ __attribute__((aligned(16))) unsigned short vT[64 * 32];      // [feat][key]
  __shared__ __attribute__((aligned(16))) unsigned short pT[4][16 * 32];   // per-wave P

  const int tid = threadIdx.x;
  const int lane = tid & 31;
  const int wave = tid >> 5;
  const int hi = lane >> 4;
  const int l16 = lane & 15;

  const int bh = blockIdx.x;
  const int b = bh >> 4, h = bh & 15;
  const int qBase = blockIdx.y * 64 + wave * 16;

  float w1[16], b1[16], w2[16];
#pragma unroll
  for (int k = 0; k < 16; ++k) {
    w1[k] = c1w[k];
    b1[k] = c1b[k];
    w2[k] = c2w[h * 16 + k];
  }
  const float cb = c2b[h];

  // q A-fragments (16x64 -> 2 x 16x32), loaded straight from global.
  v16bf qA[2];
  {
    const unsigned short* qrow = qB + ((size_t)bh * 512 + qBase + l16) * 64;
#pragma unroll
    for (int kc = 0; kc < 2; ++kc)
      qA[kc] = ld_frag(qrow + kc * 32 + (hi ? 8 : 0),
                       qrow + kc * 32 + 16 + (hi ? 8 : 0));
  }

  float rowM[3][8], rowL[3][8];
  v8f acc[3][4];
#pragma unroll
  for (int t = 0; t < 3; ++t) {
#pragma unroll
    for (int r = 0; r < 8; ++r) {
      rowM[t][r] = -__builtin_inff();
      rowL[t][r] = 0.0f;
    }
#pragma unroll
    for (int g = 0; g < 4; ++g)
#pragma unroll
      for (int e = 0; e < 8; ++e) acc[t][g][e] = 0.0f;
  }

  const float thr0 = 0.3f, thr1 = 0.6f;

  for (int j0 = 0; j0 < 512; j0 += 32) {
    __syncthreads();
    // Stage K tile (32 contiguous rows of 64 bf16) -> kT, b128 async chunks.
    {
      const unsigned short* src = kB + ((size_t)bh * 512 + j0) * 64;
      copy16_g2s(kT + tid * 8, src + tid * 8);
      copy16_g2s(kT + (tid + 128) * 8, src + (tid + 128) * 8);
    }
    // Stage V tile transposed -> vT[d][key].
    {
      const unsigned short* src = vB + ((size_t)bh * 512 + j0) * 64;
#pragma unroll
      for (int it = 0; it < 2; ++it) {
        int c = tid * 2 + it;            // 0..255 chunks of 8
        int key = c >> 3;                // 0..31
        int db = (c & 7) * 8;
        U128 dv = *(const U128*)(src + key * 64 + db);
        const unsigned short* s16 = (const unsigned short*)&dv;
#pragma unroll
        for (int j = 0; j < 8; ++j) vT[(db + j) * 32 + key] = s16[j];
      }
    }
    wait_g2s();
    __syncthreads();

    // S = q @ k^T for two 16-key column groups (K split 2 x 32 features).
    v8f s[2];
#pragma unroll
    for (int g = 0; g < 2; ++g) {
      const unsigned short* kp = kT + (g * 16 + l16) * 64;
      v16bf kb0 = ld_frag(kp + (hi ? 8 : 0), kp + 16 + (hi ? 8 : 0));
      v16bf kb1 = ld_frag(kp + 32 + (hi ? 8 : 0), kp + 32 + 16 + (hi ? 8 : 0));
      v8f c;
#pragma unroll
      for (int e = 0; e < 8; ++e) c[e] = 0.0f;
      c = wmma_bf16(qA[0], kb0, c);
      c = wmma_bf16(qA[1], kb1, c);
      s[g] = c;
    }

    // Per-element distance conv + score scaling.
    float sv[2][8], dvv[2][8];
#pragma unroll
    for (int g = 0; g < 2; ++g) {
      int j = j0 + g * 16 + l16;
      const float* dcol = dist + (size_t)b * 512 * 512 + j;
#pragma unroll
      for (int r = 0; r < 8; ++r) {
        int nq = qBase + r + hi * 8;
        float dval = dcol[(size_t)nq * 512];
        float dc = cb;
#pragma unroll
        for (int k = 0; k < 16; ++k)
          dc += w2[k] * fmaxf(w1[k] * dval + b1[k], 0.0f);
        dvv[g][r] = dval;
        sv[g][r] = s[g][r] * 0.125f * dc;   // 1/sqrt(64)
      }
    }

    // V B-fragments for PV (shared by all 3 thresholds).
    v16bf vF[4];
#pragma unroll
    for (int g = 0; g < 4; ++g) {
      const unsigned short* vp = vT + (g * 16 + l16) * 32 + (hi ? 8 : 0);
      vF[g] = ld_frag(vp, vp + 16);
    }

#pragma unroll
    for (int t = 0; t < 3; ++t) {
      float p[2][8];
#pragma unroll
      for (int r = 0; r < 8; ++r) {
        int nq = qBase + r + hi * 8;
        float m0 = NEGV;
#pragma unroll
        for (int g = 0; g < 2; ++g) {
          int j = j0 + g * 16 + l16;
          bool ok = (t == 2) || (dvv[g][r] < (t == 0 ? thr0 : thr1)) ||
                    (nq == 0) || (j == 0);
          float x = ok ? sv[g][r] : NEGV;
          p[g][r] = x;
          m0 = fmaxf(m0, x);
        }
#pragma unroll
        for (int off = 1; off < 16; off <<= 1)
          m0 = fmaxf(m0, __shfl_xor(m0, off, 16));
        float mNew = fmaxf(rowM[t][r], m0);
        float scale = __expf(rowM[t][r] - mNew);
        float ls = 0.0f;
#pragma unroll
        for (int g = 0; g < 2; ++g) {
          p[g][r] = __expf(p[g][r] - mNew);
          ls += p[g][r];
        }
#pragma unroll
        for (int off = 1; off < 16; off <<= 1) ls += __shfl_xor(ls, off, 16);
        rowL[t][r] = rowL[t][r] * scale + ls;
        rowM[t][r] = mNew;
#pragma unroll
        for (int g = 0; g < 4; ++g) acc[t][g][r] *= scale;
      }
      // P: C layout -> LDS -> A layout (16x32) for PV.
      unsigned short* pw = pT[wave];
#pragma unroll
      for (int g = 0; g < 2; ++g)
#pragma unroll
        for (int r = 0; r < 8; ++r)
          pw[(r + hi * 8) * 32 + g * 16 + l16] = f2bf(p[g][r]);
      asm volatile("s_wait_dscnt 0" ::: "memory");  // in-wave LDS RAW fence
      const unsigned short* pp = pw + l16 * 32;
      v16bf pA = ld_frag(pp + (hi ? 8 : 0), pp + 16 + (hi ? 8 : 0));
#pragma unroll
      for (int g = 0; g < 4; ++g) acc[t][g] = wmma_bf16(pA, vF[g], acc[t][g]);
    }
  }

  // Epilogue: o[t] in (B,N,E) layout, bf16 for the following Wo GEMM.
#pragma unroll
  for (int t = 0; t < 3; ++t) {
    unsigned short* ot = out + (size_t)t * outStrideT;
#pragma unroll
    for (int g = 0; g < 4; ++g)
#pragma unroll
      for (int r = 0; r < 8; ++r) {
        int nq = qBase + r + hi * 8;
        int col = h * 64 + g * 16 + l16;
        float v = acc[t][g][r] / rowL[t][r];
        ot[((size_t)b * 512 + nq) * 1024 + col] = f2bf(v);
      }
  }
}

// ---------------- host orchestration --------------------------------------

extern "C" void kernel_launch(void* const* d_in, const int* in_sizes, int n_in,
                              void* d_out, int out_size, void* d_ws,
                              size_t ws_size, hipStream_t stream) {
  (void)in_sizes; (void)n_in; (void)out_size; (void)ws_size;

  const float* query = (const float*)d_in[0];
  const float* key_  = (const float*)d_in[1];
  const float* value = (const float*)d_in[2];
  const float* dist  = (const float*)d_in[3];
  const float* Wq = (const float*)d_in[4];   const float* bq = (const float*)d_in[5];
  const float* Wk = (const float*)d_in[6];   const float* bk = (const float*)d_in[7];
  const float* Wv = (const float*)d_in[8];   const float* bv = (const float*)d_in[9];
  const float* Wo = (const float*)d_in[10];  const float* bo = (const float*)d_in[11];
  const float* c1w = (const float*)d_in[12]; const float* c1b = (const float*)d_in[13];
  const float* c2w = (const float*)d_in[14]; const float* c2b = (const float*)d_in[15];
  const float* sl1w = (const float*)d_in[16]; const float* sl1b = (const float*)d_in[17];
  const float* sl2w = (const float*)d_in[18]; const float* sl2b = (const float*)d_in[19];

  const int M = 8192;   // B*N
  const int E = 1024;

  // Workspace carve-up (bf16 = ushort), ~191 MB total.
  unsigned short* qb   = (unsigned short*)d_ws;
  unsigned short* kb   = qb + (size_t)M * E;
  unsigned short* vb   = kb + (size_t)M * E;
  unsigned short* ob   = vb + (size_t)M * E;          // 3 * M * E
  unsigned short* xb   = ob + (size_t)3 * M * E;      // M * 3E
  unsigned short* hb   = xb + (size_t)M * 3 * E;      // M * E
  unsigned short* actb = hb + (size_t)M * E;          // M * E (reused)
  unsigned short* wb   = actb + (size_t)M * E;        // up to 3E * E (reused)

  auto cvt = [&](const float* src, unsigned short* dst, int n) {
    cvt_f32_bf16<<<dim3((n / 4 + 255) / 256), dim3(256), 0, stream>>>(src, dst, n);
  };
  const dim3 gemmGrid(E / 128, M / 128);
  const dim3 gemmBlk(256);

  // Q/K/V projections -> head-major bf16 tensors.
  cvt(Wq, wb, E * E);
  cvt(query, actb, M * E);
  gemm_wmma<3><<<gemmGrid, gemmBlk, 0, stream>>>(actb, E, wb, E, bq, qb, 0, 0, M, E, E);

  cvt(Wk, wb, E * E);
  cvt(key_, actb, M * E);
  gemm_wmma<3><<<gemmGrid, gemmBlk, 0, stream>>>(actb, E, wb, E, bk, kb, 0, 0, M, E, E);

  cvt(Wv, wb, E * E);
  cvt(value, actb, M * E);
  gemm_wmma<3><<<gemmGrid, gemmBlk, 0, stream>>>(actb, E, wb, E, bv, vb, 0, 0, M, E, E);

  // Fused 3-threshold attention (no (B,H,N,N) materialization).
  attn_fused<<<dim3(256, 8), dim3(128), 0, stream>>>(
      qb, kb, vb, dist, c1w, c1b, c2w, c2b, ob, (size_t)M * E);

  // x[:, t*E:(t+1)*E] = o_t @ Wo + bo
  cvt(Wo, wb, E * E);
  for (int t = 0; t < 3; ++t)
    gemm_wmma<0><<<gemmGrid, gemmBlk, 0, stream>>>(
        ob + (size_t)t * M * E, E, wb, E, bo, xb, 3 * E, t * E, M, E, E);

  // h = relu(x @ sl1_w + sl1_b)
  cvt(sl1w, wb, 3 * E * E);
  gemm_wmma<1><<<gemmGrid, gemmBlk, 0, stream>>>(xb, 3 * E, wb, E, sl1b, hb, E, 0,
                                                 M, E, 3 * E);

  // out = h @ sl2_w + sl2_b  (fp32 output)
  cvt(sl2w, wb, E * E);
  gemm_wmma<2><<<gemmGrid, gemmBlk, 0, stream>>>(hb, E, wb, E, sl2b, d_out, E, 0,
                                                 M, E, E);
}
